// TransitionDown_687194767468
// MI455X (gfx1250) — matrix operand order repaired and checked by reading, not compiled, not run on hardware
//
#include <hip/hip_runtime.h>
#include <math.h>

#define BATCH   8
#define NFULL   4097
#define NR      4096
#define DFEAT   61
#define NPOINT  1024
#define NSAMPLE 16
#define EPSV    1e-5f
#define OUT_XYZ_ELEMS (BATCH * 1025 * 3)

typedef _Float16 v16h __attribute__((ext_vector_type(16)));
typedef _Float16 v8h  __attribute__((ext_vector_type(8)));
typedef float    v8f  __attribute__((ext_vector_type(8)));

// ---- WMMA fragment loaders (LDS, f16) ------------------------------------
// A-matrix 16x32 f16: lane l holds row m=l&15; a[0..7]=x[m][kb..kb+7],
// a[8..15]=x[m][kb+16..kb+23], kb = 32*q + (l>>4)*8. Gap between the two
// 8-half groups is 16 halves within the same row.
__device__ inline v16h ldA16(const _Float16* p) {
  v8h lo = *(const v8h*)p;
  v8h hi = *(const v8h*)(p + 16);
  v16h r;
#pragma unroll
  for (int i = 0; i < 8; ++i) { r[i] = lo[i]; r[8 + i] = hi[i]; }
  return r;
}
// B-matrix 32x16 f16: lane l holds column n=l&15; b[0..7] = W[n][kb..kb+7],
// b[8..15] = W[n][kb+16..kb+23], kb = 32*q + (l>>4)*8.
__device__ inline v16h ldB16(const _Float16* p) {
  v8h lo = *(const v8h*)p;
  v8h hi = *(const v8h*)(p + 16);
  v16h r;
#pragma unroll
  for (int i = 0; i < 8; ++i) { r[i] = lo[i]; r[8 + i] = hi[i]; }
  return r;
}

// ---- Kernel 1: cls branch (first point through 2-layer MLP, scalar) ------
__global__ __launch_bounds__(128) void cls_kernel(
    const float* __restrict__ xyz, const float* __restrict__ pts,
    const float* __restrict__ w0, const float* __restrict__ b0,
    const float* __restrict__ g0, const float* __restrict__ bt0,
    const float* __restrict__ m0, const float* __restrict__ v0,
    const float* __restrict__ w1, const float* __restrict__ b1,
    const float* __restrict__ g1, const float* __restrict__ bt1,
    const float* __restrict__ m1, const float* __restrict__ v1,
    float* __restrict__ out)
{
  __shared__ float xin[BATCH][64];
  __shared__ float h[BATCH][128];
  const int t = threadIdx.x;

  for (int i = t; i < BATCH * 64; i += 128) {
    int b = i >> 6, c = i & 63;
    xin[b][c] = (c < 3) ? xyz[b * NFULL * 3 + c] : pts[b * NFULL * DFEAT + (c - 3)];
  }
  __syncthreads();
  {
    float sc = g0[t] * rsqrtf(v0[t] + EPSV);
    float of = (b0[t] - m0[t]) * sc + bt0[t];
    for (int b = 0; b < BATCH; ++b) {
      float acc = 0.f;
#pragma unroll 8
      for (int c = 0; c < 64; ++c) acc += w0[t * 64 + c] * xin[b][c];
      h[b][t] = fmaxf(acc * sc + of, 0.f);
    }
  }
  __syncthreads();
  {
    float sc = g1[t] * rsqrtf(v1[t] + EPSV);
    float of = (b1[t] - m1[t]) * sc + bt1[t];
    float* out_pts = out + OUT_XYZ_ELEMS;
    for (int b = 0; b < BATCH; ++b) {
      float acc = 0.f;
#pragma unroll 8
      for (int c = 0; c < 128; ++c) acc += w1[t * 128 + c] * h[b][c];
      out_pts[b * 1025 * 128 + t] = fmaxf(acc * sc + of, 0.f);
    }
  }
  if (t < BATCH * 3) {
    int b = t / 3, c = t % 3;
    out[b * 1025 * 3 + c] = xyz[b * NFULL * 3 + c];
  }
}

// ---- Kernel 2: farthest-point sampling (one block per batch) -------------
__global__ __launch_bounds__(256) void fps_kernel(
    const float* __restrict__ xyz, float* __restrict__ nxyz,
    float* __restrict__ out)
{
  __shared__ float sx[NR], sy[NR], sz[NR];
  __shared__ float rmax[256];
  __shared__ int   rarg[256];
  __shared__ int   sfar;
  const int t = threadIdx.x, b = blockIdx.x;
  const float* base = xyz + b * NFULL * 3 + 3;  // xyz_r

  float px[16], py[16], pz[16], dloc[16];
#pragma unroll
  for (int j = 0; j < 16; ++j) {
    int i = t + 256 * j;
    float x = base[i * 3 + 0], y = base[i * 3 + 1], z = base[i * 3 + 2];
    px[j] = x; py[j] = y; pz[j] = z;
    sx[i] = x; sy[i] = y; sz[i] = z;
    dloc[j] = 1e10f;
  }
  __syncthreads();

  int far = 0;
  for (int step = 0; step < NPOINT; ++step) {
    float cx = sx[far], cy = sy[far], cz = sz[far];
    if (t == 0) {
      nxyz[(b * NPOINT + step) * 3 + 0] = cx;
      nxyz[(b * NPOINT + step) * 3 + 1] = cy;
      nxyz[(b * NPOINT + step) * 3 + 2] = cz;
      out[b * 1025 * 3 + (1 + step) * 3 + 0] = cx;
      out[b * 1025 * 3 + (1 + step) * 3 + 1] = cy;
      out[b * 1025 * 3 + (1 + step) * 3 + 2] = cz;
    }
    float best = -1.f; int bi = 0;
#pragma unroll
    for (int j = 0; j < 16; ++j) {
      float dx = px[j] - cx, dy = py[j] - cy, dz = pz[j] - cz;
      float d  = dx * dx + dy * dy + dz * dz;
      float nd = fminf(dloc[j], d);
      dloc[j]  = nd;
      if (nd > best) { best = nd; bi = t + 256 * j; }
    }
    rmax[t] = best; rarg[t] = bi;
    __syncthreads();
    for (int sft = 128; sft > 0; sft >>= 1) {
      if (t < sft) {
        float om = rmax[t + sft]; int oa = rarg[t + sft];
        if (om > rmax[t] || (om == rmax[t] && oa < rarg[t])) { rmax[t] = om; rarg[t] = oa; }
      }
      __syncthreads();
    }
    if (t == 0) sfar = rarg[0];
    __syncthreads();
    far = sfar;
  }
}

// ---- Kernel 3: 16-NN per sampled point (one block per query) -------------
__global__ __launch_bounds__(256) void knn_kernel(
    const float* __restrict__ xyz, const float* __restrict__ nxyz,
    int* __restrict__ knn)
{
  __shared__ float dist[NR];
  __shared__ float rmin[256];
  __shared__ int   rarg[256];
  const int t = threadIdx.x;
  const int q = blockIdx.x;            // 0..8191
  const int b = q >> 10;
  const float qx = nxyz[q * 3 + 0], qy = nxyz[q * 3 + 1], qz = nxyz[q * 3 + 2];
  const float* base = xyz + b * NFULL * 3 + 3;

  for (int i = t; i < NR; i += 256) {
    float dx = base[i * 3 + 0] - qx, dy = base[i * 3 + 1] - qy, dz = base[i * 3 + 2] - qz;
    dist[i] = dx * dx + dy * dy + dz * dz;
  }
  __syncthreads();

  for (int k = 0; k < NSAMPLE; ++k) {
    float best = 1e30f; int bi = 0;
#pragma unroll
    for (int j = 0; j < 16; ++j) {
      int i = t + 256 * j;
      float d = dist[i];
      if (d < best) { best = d; bi = i; }
    }
    rmin[t] = best; rarg[t] = bi;
    __syncthreads();
    for (int sft = 128; sft > 0; sft >>= 1) {
      if (t < sft) {
        float om = rmin[t + sft]; int oa = rarg[t + sft];
        if (om < rmin[t] || (om == rmin[t] && oa < rarg[t])) { rmin[t] = om; rarg[t] = oa; }
      }
      __syncthreads();
    }
    int sel = rarg[0];
    if (t == 0) knn[q * NSAMPLE + k] = sel;
    if ((sel & 255) == t) dist[sel] = 1e30f;   // owner thread marks it used
    __syncthreads();
  }
}

// ---- Kernel 4: grouped 2-layer MLP + max-pool via WMMA -------------------
// 2 waves / block, 1 query (16 samples x 64 feats) per wave.
__global__ __launch_bounds__(64) void mlp_kernel(
    const float* __restrict__ xyz, const float* __restrict__ pts,
    const float* __restrict__ nxyz, const int* __restrict__ knn,
    const float* __restrict__ w0, const float* __restrict__ b0,
    const float* __restrict__ g0, const float* __restrict__ bt0,
    const float* __restrict__ m0, const float* __restrict__ v0,
    const float* __restrict__ w1, const float* __restrict__ b1,
    const float* __restrict__ g1, const float* __restrict__ bt1,
    const float* __restrict__ m1, const float* __restrict__ v1,
    float* __restrict__ out)
{
  __shared__ __align__(16) _Float16 w0h[128][64];     // 16 KB
  __shared__ __align__(16) _Float16 w1h[128][128];    // 32 KB
  __shared__ float bn0s[128], bn0t[128], bn1s[128], bn1t[128];  // 2 KB
  __shared__ __align__(16) _Float16 act0[2][16][64];  // 4 KB
  __shared__ __align__(16) _Float16 act1[2][16][128]; // 8 KB

  const int tid = threadIdx.x;

  // Stage weights as f16 and folded BN params.
  {
    _Float16* w0f = &w0h[0][0];
    for (int i = tid; i < 128 * 64; i += 64)  w0f[i] = (_Float16)w0[i];
    _Float16* w1f = &w1h[0][0];
    for (int i = tid; i < 128 * 128; i += 64) w1f[i] = (_Float16)w1[i];
    for (int o = tid; o < 128; o += 64) {
      float sc0 = g0[o] * rsqrtf(v0[o] + EPSV);
      bn0s[o] = sc0; bn0t[o] = (b0[o] - m0[o]) * sc0 + bt0[o];
      float sc1 = g1[o] * rsqrtf(v1[o] + EPSV);
      bn1s[o] = sc1; bn1t[o] = (b1[o] - m1[o]) * sc1 + bt1[o];
    }
  }
  __syncthreads();

  const int wv = tid >> 5, lane = tid & 31;
  const int q = blockIdx.x * 2 + wv;
  const int b = q >> 10, sq = q & 1023;
  const float qx = nxyz[q * 3 + 0], qy = nxyz[q * 3 + 1], qz = nxyz[q * 3 + 2];
  const int* kn = knn + q * NSAMPLE;

  // Gather 16 neighbor rows of 64 features (rel-xyz ++ point feats) as f16.
  for (int e = lane; e < NSAMPLE * 64; e += 32) {
    int m = e >> 6, c = e & 63;
    int idx = kn[m];
    float v;
    if (c < 3) {
      float qc = (c == 0) ? qx : ((c == 1) ? qy : qz);
      v = xyz[b * NFULL * 3 + (idx + 1) * 3 + c] - qc;
    } else {
      v = pts[b * NFULL * DFEAT + (idx + 1) * DFEAT + (c - 3)];
    }
    act0[wv][m][c] = (_Float16)v;
  }
  __syncthreads();

  const int mrow = lane & 15, hf = lane >> 4;

  // Layer 0: (16x64) x (64x128) -> 16x128, BN+ReLU, restage as f16.
  v16h A0 = ldA16(&act0[wv][mrow][hf * 8]);
  v16h A1 = ldA16(&act0[wv][mrow][32 + hf * 8]);
#pragma unroll
  for (int tN = 0; tN < 8; ++tN) {
    int n = tN * 16 + mrow;
    v16h B0 = ldB16(&w0h[n][hf * 8]);
    v16h B1 = ldB16(&w0h[n][32 + hf * 8]);
    v8f c = {0.f, 0.f, 0.f, 0.f, 0.f, 0.f, 0.f, 0.f};
    c = __builtin_amdgcn_wmma_f32_16x16x32_f16(false, A0, false, B0, (short)0, c, false, false);
    c = __builtin_amdgcn_wmma_f32_16x16x32_f16(false, A1, false, B1, (short)0, c, false, false);
    float sc = bn0s[n], of = bn0t[n];
#pragma unroll
    for (int r = 0; r < 8; ++r) {
      float v = fmaxf(c[r] * sc + of, 0.f);
      act1[wv][r + hf * 8][n] = (_Float16)v;
    }
  }
  __syncthreads();

  // Layer 1: (16x128) x (128x128) -> 16x128, BN+ReLU, max over 16 samples.
  v16h A[4];
#pragma unroll
  for (int qk = 0; qk < 4; ++qk)
    A[qk] = ldA16(&act1[wv][mrow][qk * 32 + hf * 8]);

  float* out_pts = out + OUT_XYZ_ELEMS;
#pragma unroll
  for (int tN = 0; tN < 8; ++tN) {
    int n = tN * 16 + mrow;
    v8f c = {0.f, 0.f, 0.f, 0.f, 0.f, 0.f, 0.f, 0.f};
#pragma unroll
    for (int qk = 0; qk < 4; ++qk) {
      v16h Bq = ldB16(&w1h[n][qk * 32 + hf * 8]);
      c = __builtin_amdgcn_wmma_f32_16x16x32_f16(false, A[qk], false, Bq, (short)0, c, false, false);
    }
    float sc = bn1s[n], of = bn1t[n];
    float mx = -1e30f;
#pragma unroll
    for (int r = 0; r < 8; ++r)
      mx = fmaxf(mx, fmaxf(c[r] * sc + of, 0.f));
    float other = __shfl_xor(mx, 16, 32);   // combine the two row-halves
    mx = fmaxf(mx, other);
    if (lane < 16)
      out_pts[(b * 1025 + 1 + sq) * 128 + n] = mx;
  }
}

// ---- Host launch ---------------------------------------------------------
extern "C" void kernel_launch(void* const* d_in, const int* in_sizes, int n_in,
                              void* d_out, int out_size, void* d_ws, size_t ws_size,
                              hipStream_t stream)
{
  const float* xyz = (const float*)d_in[0];
  const float* pts = (const float*)d_in[1];
  // sa layer0: 2..7, cls layer0: 8..13, sa layer1: 14..19, cls layer1: 20..25
  const float* sa_w0 = (const float*)d_in[2];
  const float* sa_b0 = (const float*)d_in[3];
  const float* sa_g0 = (const float*)d_in[4];
  const float* sa_bt0 = (const float*)d_in[5];
  const float* sa_m0 = (const float*)d_in[6];
  const float* sa_v0 = (const float*)d_in[7];
  const float* cls_w0 = (const float*)d_in[8];
  const float* cls_b0 = (const float*)d_in[9];
  const float* cls_g0 = (const float*)d_in[10];
  const float* cls_bt0 = (const float*)d_in[11];
  const float* cls_m0 = (const float*)d_in[12];
  const float* cls_v0 = (const float*)d_in[13];
  const float* sa_w1 = (const float*)d_in[14];
  const float* sa_b1 = (const float*)d_in[15];
  const float* sa_g1 = (const float*)d_in[16];
  const float* sa_bt1 = (const float*)d_in[17];
  const float* sa_m1 = (const float*)d_in[18];
  const float* sa_v1 = (const float*)d_in[19];
  const float* cls_w1 = (const float*)d_in[20];
  const float* cls_b1 = (const float*)d_in[21];
  const float* cls_g1 = (const float*)d_in[22];
  const float* cls_bt1 = (const float*)d_in[23];
  const float* cls_m1 = (const float*)d_in[24];
  const float* cls_v1 = (const float*)d_in[25];

  float* out  = (float*)d_out;
  float* nxyz = (float*)d_ws;                                           // B*1024*3 f32
  int*   knn  = (int*)((char*)d_ws + (size_t)BATCH * NPOINT * 3 * sizeof(float));

  cls_kernel<<<1, 128, 0, stream>>>(xyz, pts,
      cls_w0, cls_b0, cls_g0, cls_bt0, cls_m0, cls_v0,
      cls_w1, cls_b1, cls_g1, cls_bt1, cls_m1, cls_v1, out);

  fps_kernel<<<BATCH, 256, 0, stream>>>(xyz, nxyz, out);

  knn_kernel<<<BATCH * NPOINT, 256, 0, stream>>>(xyz, nxyz, knn);

  mlp_kernel<<<BATCH * NPOINT / 2, 64, 0, stream>>>(xyz, pts, nxyz, knn,
      sa_w0, sa_b0, sa_g0, sa_bt0, sa_m0, sa_v0,
      sa_w1, sa_b1, sa_g1, sa_bt1, sa_m1, sa_v1, out);
}
